// RnnCommon_24927990186129
// MI455X (gfx1250) — compile-verified, matrix operand order checked by reading, not run on hardware
//
#include <hip/hip_runtime.h>

// ---------------------------------------------------------------------------
// Bidirectional 2-layer LSTM, B=64 T=512 D=H=768, for MI455X (gfx1250).
//  * bf16 WMMA 16x16x32, fp32 accumulate; cell state c stays fp32.
//  * Gates = [x_t, h_{t-1}] @ [W_ih; W_hh]^T  (one GEMM, concatenated K),
//    split into an x-tile loop and an h-tile loop (branch-free inner loops).
//  * Weights pre-packed into WMMA B-fragment order (47MB bf16 -> L2 resident).
//  * Persistent kernel: 24 WGs (2 dirs x 12), each owns 64 hidden units
//    (256 permuted gate columns). Device-scope atomic generation barrier
//    per timestep; h fragments double-buffered in global memory.
//  * global_prefetch of next timestep's x fragments (1 cacheline/lane/ktile).
// ---------------------------------------------------------------------------

#define TT      512
#define BB      64
#define HH      768
#define NWG_DIR 12
#define NB      24          // total persistent workgroups (2 dirs * 12)

typedef __attribute__((ext_vector_type(16))) __bf16 v16bf;
typedef __attribute__((ext_vector_type(8)))  float  v8f;

__device__ __forceinline__ unsigned bf16r(float f) {
  unsigned u = __float_as_uint(f);
  return (u + 0x7FFFu + ((u >> 16) & 1u)) >> 16;   // round-to-nearest-even
}
__device__ __forceinline__ unsigned pack2bf(float a, float b) {
  return bf16r(a) | (bf16r(b) << 16);
}
// K index held by (dword v, lane-half) in a 16-bit A/B WMMA fragment
__device__ __forceinline__ int klane(int v, int half) {
  return ((v < 4) ? (2 * v) : (16 + 2 * (v - 4))) + 8 * half;
}

// ------------------------- packing kernels ---------------------------------

// inputs [B,T,768] fp32 -> A-fragments [T][24 kt][4 mt][32 lane][16 bf16]
__global__ __launch_bounds__(256) void pack_x0(const float* __restrict__ x,
                                               unsigned short* __restrict__ dst) {
  unsigned gid = blockIdx.x * 256 + threadIdx.x;   // T*24*4*32 threads exactly
  int lane = gid & 31;
  int mt   = (gid >> 5) & 3;
  int kt   = (gid >> 7) % 24;
  int t    = (gid >> 7) / 24;
  int m = lane & 15, half = lane >> 4;
  int b = mt * 16 + m;
  unsigned dw[8];
#pragma unroll
  for (int v = 0; v < 8; ++v) {
    int k = kt * 32 + klane(v, half);
    float f0 = x[((size_t)b * TT + t) * 768 + k];
    float f1 = x[((size_t)b * TT + t) * 768 + k + 1];
    dw[v] = pack2bf(f0, f1);
  }
  unsigned short* p = dst + ((((size_t)t * 24 + kt) * 4 + mt) * 512) + lane * 16;
  ((uint4*)p)[0] = make_uint4(dw[0], dw[1], dw[2], dw[3]);
  ((uint4*)p)[1] = make_uint4(dw[4], dw[5], dw[6], dw[7]);
}

// [W_ih; W_hh]^T -> B-fragments, gate columns permuted so each WG's 256
// columns are contiguous:  pc -> row = q*768 + wg*64 + jj  (q = i/f/g/o)
template <int KT, int IN>
__global__ __launch_bounds__(256) void pack_w(const float* __restrict__ wih_f,
                                              const float* __restrict__ whh_f,
                                              const float* __restrict__ wih_b,
                                              const float* __restrict__ whh_b,
                                              unsigned short* __restrict__ dst) {
  unsigned gid = blockIdx.x * 256 + threadIdx.x;   // 2*12*16*KT*32 exactly
  int lane = gid & 31;
  unsigned r1 = gid >> 5;
  int kt  = r1 % KT;  r1 /= KT;
  int ntL = r1 & 15;  r1 >>= 4;
  int wg  = r1 % NWG_DIR;
  int dir = r1 / NWG_DIR;
  const float* wih = dir ? wih_b : wih_f;
  const float* whh = dir ? whh_b : whh_f;
  int nl = lane & 15, half = lane >> 4;
  int pcl = ntL * 16 + nl;                 // 0..255 local column
  int q = pcl >> 6, jj = pcl & 63;
  int row = q * HH + wg * 64 + jj;         // original weight row (gate unit)
  unsigned dw[8];
#pragma unroll
  for (int v = 0; v < 8; ++v) {
    int k = kt * 32 + klane(v, half);      // kTile never straddles IN (both %32==0)
    float f0, f1;
    if (k < IN) { f0 = wih[(size_t)row * IN + k];        f1 = wih[(size_t)row * IN + k + 1]; }
    else        { f0 = whh[(size_t)row * HH + (k - IN)]; f1 = whh[(size_t)row * HH + (k - IN) + 1]; }
    dw[v] = pack2bf(f0, f1);
  }
  unsigned short* p = dst +
      (((((size_t)dir * NWG_DIR + wg) * 16 + ntL) * KT + kt) * 512) + lane * 16;
  ((uint4*)p)[0] = make_uint4(dw[0], dw[1], dw[2], dw[3]);
  ((uint4*)p)[1] = make_uint4(dw[4], dw[5], dw[6], dw[7]);
}

__global__ __launch_bounds__(256) void pack_bias(const float* __restrict__ bi_f,
                                                 const float* __restrict__ bh_f,
                                                 const float* __restrict__ bi_b,
                                                 const float* __restrict__ bh_b,
                                                 float* __restrict__ dst) {
  int gid = blockIdx.x * 256 + threadIdx.x;          // 2*3072 exactly
  int dir = gid / 3072, pc = gid % 3072;
  int wg = pc >> 8, q = (pc >> 6) & 3, jj = pc & 63;
  int row = q * HH + wg * 64 + jj;
  const float* bi = dir ? bi_b : bi_f;
  const float* bh = dir ? bh_b : bh_f;
  dst[gid] = bi[row] + bh[row];
}

__global__ __launch_bounds__(256) void init_zero(unsigned* __restrict__ hp, unsigned n,
                                                 unsigned* __restrict__ ctr) {
  unsigned i = blockIdx.x * 256 + threadIdx.x;
  if (i < n) hp[i] = 0u;
  if (i == 0) *ctr = 0u;
}

// ------------------------- persistent LSTM layer ---------------------------

#define GSTRIDE 260   // 64x256 gate tile, padded vs 64 LDS banks
#define HSTRIDE 67    // 64x64 h tile, padded

template <int IN_TILES, int KT, int LAYER>
__global__ __launch_bounds__(256) void lstm_layer(
    const unsigned short* __restrict__ xfrag,   // [T][IN_TILES][4][512]
    const unsigned short* __restrict__ wfrag,   // [2][12][16][KT][512]
    const float* __restrict__ biasP,            // [2][3072] permuted
    const float* __restrict__ mask,             // [B][T]
    unsigned short* __restrict__ hPack,         // [2*2][24][4][512] dbl-buffered
    unsigned short* __restrict__ yfragNext,     // layer0: packed input of layer1
    float* __restrict__ out,                    // [B][T][1536] (layer1 only)
    float* __restrict__ hn, float* __restrict__ cn,  // [B][1536], layer slice
    unsigned* __restrict__ ctr) {
  constexpr int H_TILES = KT - IN_TILES;       // 24 (recurrent K-tiles)
  __shared__ float gates[BB * GSTRIDE];
  __shared__ float hbuf[BB * HSTRIDE];

  const int tid = threadIdx.x, lane = tid & 31, wave = tid >> 5;
  const int dir = blockIdx.x / NWG_DIR, wg = blockIdx.x % NWG_DIR;

  float c[16];
#pragma unroll
  for (int r = 0; r < 16; ++r) c[r] = 0.f;

  // B-fragment bases for this wave's two local N-tiles
  const unsigned short* wbase0 =
      wfrag + ((((size_t)dir * NWG_DIR + wg) * 16 + 2 * wave) * KT) * 512 + lane * 16;
  const unsigned short* wbase1 = wbase0 + (size_t)KT * 512;

  float bsel[4];
  {
    int jj = tid & 63;
#pragma unroll
    for (int q = 0; q < 4; ++q)
      bsel[q] = biasP[dir * 3072 + wg * 256 + q * 64 + jj];
  }

  for (int s = 0; s < TT; ++s) {
    const int t = dir ? (TT - 1 - s) : s;
    // next timestep (clamped) for activation prefetch
    const int tn = dir ? (t > 0 ? t - 1 : 0) : (t < TT - 1 ? t + 1 : TT - 1);

    v8f acc[4][2];
#pragma unroll
    for (int mt = 0; mt < 4; ++mt)
#pragma unroll
      for (int nn = 0; nn < 2; ++nn)
        acc[mt][nn] = (v8f){0.f, 0.f, 0.f, 0.f, 0.f, 0.f, 0.f, 0.f};

    // ---- x contribution: branch-free K loop over the input tiles ----
    {
      const unsigned short* aT  = xfrag + ((size_t)t  * IN_TILES) * 4 * 512;
      const unsigned short* aTn = xfrag + ((size_t)tn * IN_TILES) * 4 * 512;
      for (int kt = 0; kt < IN_TILES; ++kt) {
        const unsigned short* aB = aT + (size_t)kt * 4 * 512;
        // pull next step's fragments toward the WGP: 4KB tile = 1 line/lane
        __builtin_prefetch(aTn + (size_t)kt * 4 * 512 + lane * 64, 0, 3);
        v16bf a[4];
#pragma unroll
        for (int mt = 0; mt < 4; ++mt)
          a[mt] = *(const v16bf*)(const void*)(aB + (size_t)mt * 512 + lane * 16);
        v16bf b0 = *(const v16bf*)(const void*)(wbase0 + (size_t)kt * 512);
        v16bf b1 = *(const v16bf*)(const void*)(wbase1 + (size_t)kt * 512);
#pragma unroll
        for (int mt = 0; mt < 4; ++mt) {
          acc[mt][0] = __builtin_amdgcn_wmma_f32_16x16x32_bf16(
              false, a[mt], false, b0, (short)0, acc[mt][0], false, false);
          acc[mt][1] = __builtin_amdgcn_wmma_f32_16x16x32_bf16(
              false, a[mt], false, b1, (short)0, acc[mt][1], false, false);
        }
      }
    }

    // ---- h contribution: branch-free K loop over the recurrent tiles ----
    {
      const unsigned short* hR =
          hPack + (((size_t)dir * 2 + (s & 1)) * 24) * 4 * 512;
      const unsigned short* wh0 = wbase0 + (size_t)IN_TILES * 512;
      const unsigned short* wh1 = wbase1 + (size_t)IN_TILES * 512;
      for (int kt = 0; kt < H_TILES; ++kt) {
        const unsigned short* aB = hR + (size_t)kt * 4 * 512;
        v16bf a[4];
#pragma unroll
        for (int mt = 0; mt < 4; ++mt)
          a[mt] = *(const v16bf*)(const void*)(aB + (size_t)mt * 512 + lane * 16);
        v16bf b0 = *(const v16bf*)(const void*)(wh0 + (size_t)kt * 512);
        v16bf b1 = *(const v16bf*)(const void*)(wh1 + (size_t)kt * 512);
#pragma unroll
        for (int mt = 0; mt < 4; ++mt) {
          acc[mt][0] = __builtin_amdgcn_wmma_f32_16x16x32_bf16(
              false, a[mt], false, b0, (short)0, acc[mt][0], false, false);
          acc[mt][1] = __builtin_amdgcn_wmma_f32_16x16x32_bf16(
              false, a[mt], false, b1, (short)0, acc[mt][1], false, false);
        }
      }
    }

    // Accumulators -> LDS gate tile. C layout: VGPR r, lanes<16: M=r, else M=r+8
    {
      const int ncol = (2 * wave) * 16 + (lane & 15);
      const int mrow = (lane >> 4) * 8;
#pragma unroll
      for (int mt = 0; mt < 4; ++mt)
#pragma unroll
        for (int r = 0; r < 8; ++r) {
          int brow = mt * 16 + mrow + r;
          gates[brow * GSTRIDE + ncol]      = acc[mt][0][r];
          gates[brow * GSTRIDE + ncol + 16] = acc[mt][1][r];
        }
    }
    __syncthreads();

    // Elementwise LSTM cell: each thread owns 16 (batch, hidden) elements
    {
      const int jj = tid & 63;
      const int b0i = tid >> 6;
#pragma unroll
      for (int r = 0; r < 16; ++r) {
        int b = b0i + 4 * r;
        float gi = gates[b * GSTRIDE + jj]        + bsel[0];
        float gf = gates[b * GSTRIDE + 64 + jj]   + bsel[1];
        float gg = gates[b * GSTRIDE + 128 + jj]  + bsel[2];
        float go = gates[b * GSTRIDE + 192 + jj]  + bsel[3];
        float si = 1.f / (1.f + expf(-gi));
        float sf = 1.f / (1.f + expf(-gf));
        float so = 1.f / (1.f + expf(-go));
        float cnew = sf * c[r] + si * tanhf(gg);
        float hnew = so * tanhf(cnew);           // h uses unmasked c (ref semantics)
        float m = mask[(size_t)b * TT + t];
        float hm = hnew * m, cm = cnew * m;
        c[r] = cm;
        hbuf[b * HSTRIDE + jj] = hm;
        if constexpr (LAYER == 1)
          out[((size_t)b * TT + t) * 1536 + dir * HH + wg * 64 + jj] = hm;
        if (s == TT - 1) {
          int col = dir * HH + wg * 64 + jj;
          hn[(size_t)b * 1536 + col] = hm;
          cn[(size_t)b * 1536 + col] = cm;
        }
      }
    }
    __syncthreads();

    // Re-pack h into A-fragment order: 256 thr = 2 kTiles x 4 mTiles x 32 lanes
    {
      const int fmt = wave & 3, fkt = wave >> 2;
      const int m = lane & 15, half = lane >> 4;
      unsigned dw[8];
#pragma unroll
      for (int v = 0; v < 8; ++v) {
        int j = fkt * 32 + klane(v, half);
        float f0 = hbuf[(fmt * 16 + m) * HSTRIDE + j];
        float f1 = hbuf[(fmt * 16 + m) * HSTRIDE + j + 1];
        dw[v] = pack2bf(f0, f1);
      }
      uint4 lo = make_uint4(dw[0], dw[1], dw[2], dw[3]);
      uint4 hi = make_uint4(dw[4], dw[5], dw[6], dw[7]);
      unsigned short* hp = hPack +
          (((((size_t)dir * 2 + ((s + 1) & 1)) * 24 + (wg * 2 + fkt)) * 4 + fmt) * 512) +
          lane * 16;
      ((uint4*)hp)[0] = lo;
      ((uint4*)hp)[1] = hi;
      if constexpr (LAYER == 0) {
        unsigned short* yp = yfragNext +
            ((((size_t)t * 48 + (dir * 24 + wg * 2 + fkt)) * 4 + fmt) * 512) +
            lane * 16;
        ((uint4*)yp)[0] = lo;
        ((uint4*)yp)[1] = hi;
      }
    }

    // Device-wide generation barrier (persistent grid)
    __threadfence();
    __syncthreads();
    if (tid == 0) {
      __hip_atomic_fetch_add(ctr, 1u, __ATOMIC_RELEASE, __HIP_MEMORY_SCOPE_AGENT);
      const unsigned target = (unsigned)NB * (unsigned)(s + 1);
      while (__hip_atomic_load(ctr, __ATOMIC_ACQUIRE, __HIP_MEMORY_SCOPE_AGENT) < target)
        __builtin_amdgcn_s_sleep(2);
    }
    __syncthreads();
    __threadfence();
  }
}

// ------------------------------ launcher -----------------------------------

extern "C" void kernel_launch(void* const* d_in, const int* in_sizes, int n_in,
                              void* d_out, int out_size, void* d_ws, size_t ws_size,
                              hipStream_t stream) {
  (void)in_sizes; (void)n_in; (void)out_size; (void)ws_size;
  const float* x    = (const float*)d_in[0];
  const float* mask = (const float*)d_in[1];
  // params flattened: per layer, per dir: w_ih, w_hh, b_ih, b_hh
  const float* wih00 = (const float*)d_in[2];
  const float* whh00 = (const float*)d_in[3];
  const float* bih00 = (const float*)d_in[4];
  const float* bhh00 = (const float*)d_in[5];
  const float* wih01 = (const float*)d_in[6];
  const float* whh01 = (const float*)d_in[7];
  const float* bih01 = (const float*)d_in[8];
  const float* bhh01 = (const float*)d_in[9];
  const float* wih10 = (const float*)d_in[10];
  const float* whh10 = (const float*)d_in[11];
  const float* bih10 = (const float*)d_in[12];
  const float* bhh10 = (const float*)d_in[13];
  const float* wih11 = (const float*)d_in[14];
  const float* whh11 = (const float*)d_in[15];
  const float* bih11 = (const float*)d_in[16];
  const float* bhh11 = (const float*)d_in[17];

  char* ws = (char*)d_ws;
  size_t off = 0;
  auto take = [&](size_t bytes) -> char* {
    char* p = ws + off;
    off = (off + bytes + 255) & ~(size_t)255;
    return p;
  };
  unsigned short* X0 = (unsigned short*)take((size_t)TT * 24 * 4 * 1024);   //  48 MB
  unsigned short* X1 = (unsigned short*)take((size_t)TT * 48 * 4 * 1024);   //  96 MB
  unsigned short* W0 = (unsigned short*)take((size_t)2 * 12 * 16 * 48 * 1024); // 18 MB
  unsigned short* W1 = (unsigned short*)take((size_t)2 * 12 * 16 * 72 * 1024); // 27 MB
  float*          B0 = (float*)take(2 * 3072 * 4);
  float*          B1 = (float*)take(2 * 3072 * 4);
  unsigned short* HP = (unsigned short*)take((size_t)2 * 2 * 24 * 4 * 1024);
  unsigned*      CTR = (unsigned*)take(256);

  float* out = (float*)d_out;
  float* hn  = out + (size_t)BB * TT * 1536;   // [2][64][1536]
  float* cn  = hn + (size_t)2 * BB * 1536;

  // pack input + weights + biases
  pack_x0<<<(TT * 24 * 4 * 32) / 256, 256, 0, stream>>>(x, X0);
  pack_w<48, 768><<<(2 * 12 * 16 * 48 * 32) / 256, 256, 0, stream>>>(
      wih00, whh00, wih01, whh01, W0);
  pack_w<72, 1536><<<(2 * 12 * 16 * 72 * 32) / 256, 256, 0, stream>>>(
      wih10, whh10, wih11, whh11, W1);
  pack_bias<<<(2 * 3072) / 256, 256, 0, stream>>>(bih00, bhh00, bih01, bhh01, B0);
  pack_bias<<<(2 * 3072) / 256, 256, 0, stream>>>(bih10, bhh10, bih11, bhh11, B1);

  const unsigned hpDw = 2u * 2u * 24u * 4u * 1024u / 4u;  // hPack dwords
  init_zero<<<(hpDw + 255) / 256, 256, 0, stream>>>((unsigned*)HP, hpDw, CTR);
  lstm_layer<24, 48, 0><<<NB, 256, 0, stream>>>(
      X0, W0, B0, mask, HP, X1, out, hn, cn, CTR);

  init_zero<<<(hpDw + 255) / 256, 256, 0, stream>>>((unsigned*)HP, hpDw, CTR);
  lstm_layer<48, 72, 1><<<NB, 256, 0, stream>>>(
      X1, W1, B1, mask, HP, nullptr, out, hn + (size_t)BB * 1536,
      cn + (size_t)BB * 1536, CTR);
}
